// Attention_22814866276679
// MI455X (gfx1250) — matrix operand order repaired and checked by reading, not compiled, not run on hardware
//
#include <hip/hip_runtime.h>

typedef __attribute__((ext_vector_type(16))) __bf16          v16bf;
typedef __attribute__((ext_vector_type(16))) unsigned short  v16us;
typedef __attribute__((ext_vector_type(8)))  float           v8f;
typedef __attribute__((ext_vector_type(4)))  unsigned int    v4u;
typedef __attribute__((ext_vector_type(4)))  int             v4i;

#define BATCH 64
#define NH    12
#define NTOK  197
#define DIM   768
#define HD    64
#define MROWS (BATCH * NTOK)      // 12608, 788 tiles of 16, 197 blocks of 64
#define MT    (MROWS / 16)        // 788
#define MB    (MT / 4)            // 197 cooperative 64-row blocks
#define NPAD  208                 // 13*16 row/col pad for q,k,bias
#define VPAD  224                 // 7*32  j pad for v / p (K-dim of P@V)
#define BH    (BATCH * NH)        // 768
#define KSTEPS (DIM / 32)         // 24

// LDS staging geometry for the cooperative B strip (128 N-rows x 32 K, bf16 hi+lo)
#define BROW   40                 // row stride in shorts (80B) - bank-rotating pad
#define BPLANE (128 * BROW)       // one plane (hi or lo) of one stage
#define BSTAGE (2 * BPLANE)       // hi + lo
// total LDS: 2 stages * BSTAGE shorts = 40960 B

#if defined(__has_builtin)
#  if __has_builtin(__builtin_amdgcn_global_load_async_to_lds_b128)
#    define HAS_ASYNC_LDS 1
#  endif
#endif
#ifndef HAS_ASYNC_LDS
#  define HAS_ASYNC_LDS 0
#endif

#if HAS_ASYNC_LDS
typedef __attribute__((address_space(1))) v4i* gv4i_p;   // global v4i*
typedef __attribute__((address_space(3))) v4i* lv4i_p;   // LDS v4i*
#endif

template <int N>
__device__ __forceinline__ void wait_async_le() {
#if HAS_ASYNC_LDS
#  if defined(__has_builtin) && __has_builtin(__builtin_amdgcn_s_wait_asynccnt)
    __builtin_amdgcn_s_wait_asynccnt(N);
#  else
    asm volatile("s_wait_asynccnt %0" :: "i"(N) : "memory");
#  endif
#endif
}

// ---------- bf16 split helpers (hi/lo decomposition for bf16x3 emulation) ----------
__device__ __forceinline__ unsigned short bf16_rne(float f) {
    unsigned u = __float_as_uint(f);
    u += 0x7FFFu + ((u >> 16) & 1u);
    return (unsigned short)(u >> 16);
}
__device__ __forceinline__ float bf16_to_f(unsigned short h) {
    return __uint_as_float(((unsigned)h) << 16);
}
__device__ __forceinline__ void splitf(float f, unsigned short& hi, unsigned short& lo) {
    hi = bf16_rne(f);
    lo = bf16_rne(f - bf16_to_f(hi));
}

// ---------- WMMA wrapper:  D(f32 16x16) = A(bf16 16x32) * B(bf16 32x16) + C ----------
__device__ __forceinline__ v8f wmma_bf16(v16us a, v16us b, v8f c) {
    return __builtin_amdgcn_wmma_f32_16x16x32_bf16(
        false, __builtin_bit_cast(v16bf, a),
        false, __builtin_bit_cast(v16bf, b),
        (short)0, c, false, false);
}

// load 16 contiguous bf16 (two 16B chunks) into a fragment register set
__device__ __forceinline__ v16us ld_frag16(const unsigned short* p0, const unsigned short* p1) {
    union { v4u u[2]; v16us v; } t;
    t.u[0] = *(const v4u*)p0;
    t.u[1] = *(const v4u*)p1;
    return t.v;
}

// ---------- elementwise f32 -> (bf16 hi, bf16 lo) planes ----------
__global__ void split_kernel(const float* __restrict__ src,
                             unsigned short* __restrict__ hi,
                             unsigned short* __restrict__ lo, int n) {
    int i = blockIdx.x * blockDim.x + threadIdx.x;
    if (i < n) {
        unsigned short h, l;
        splitf(src[i], h, l);
        hi[i] = h; lo[i] = l;
    }
}

// ---------- relative-position bias gather:  bias[h][i][j] = rel_table[rel_index[i,j], h] ----------
__global__ void biasfill_kernel(const float* __restrict__ rel_table,
                                const int* __restrict__ rel_index,
                                float* __restrict__ biasws) {
    int idx = blockIdx.x * blockDim.x + threadIdx.x;
    if (idx < NTOK * NTOK) {
        int i = idx / NTOK, j = idx - i * NTOK;
        int ri = rel_index[idx];
#pragma unroll
        for (int h = 0; h < NH; ++h)
            biasws[((size_t)h * NPAD + i) * NPAD + j] = rel_table[ri * NH + h];
    }
}

// ---------- cooperative staging of one K-slab of the B strip into LDS ----------
// 128 threads: thread t owns B-row (nbase + t); copies 4x16B per plane.
__device__ __forceinline__ void stage_B(const unsigned short* __restrict__ Bhi,
                                        const unsigned short* __restrict__ Blo,
                                        unsigned short* stage, int nbase, int kb, int tid) {
    const unsigned short* gh = Bhi + (size_t)(nbase + tid) * DIM + kb;
    const unsigned short* gl = Blo + (size_t)(nbase + tid) * DIM + kb;
    unsigned short* lh = stage + tid * BROW;
    unsigned short* ll = stage + BPLANE + tid * BROW;
#if HAS_ASYNC_LDS
#pragma unroll
    for (int i = 0; i < 4; ++i) {
        __builtin_amdgcn_global_load_async_to_lds_b128(
            (gv4i_p)(gh + i * 8), (lv4i_p)(lh + i * 8), 0, 0);
        __builtin_amdgcn_global_load_async_to_lds_b128(
            (gv4i_p)(gl + i * 8), (lv4i_p)(ll + i * 8), 0, 0);
    }
#else
#pragma unroll
    for (int i = 0; i < 4; ++i) {
        *(v4u*)(lh + i * 8) = *(const v4u*)(gh + i * 8);
        *(v4u*)(ll + i * 8) = *(const v4u*)(gl + i * 8);
    }
#endif
}

// ---------- block-cooperative GEMM: 4 waves x 16 rows, shared 128-wide B strip via LDS ----------
// A[M,768] bf16 hi/lo planes ; B[N,768] bf16 hi/lo planes ; computes C = A * B^T for this strip.
__device__ __forceinline__ void gemm_block(const unsigned short* __restrict__ Ahi,
                                           const unsigned short* __restrict__ Alo,
                                           const unsigned short* __restrict__ Bhi,
                                           const unsigned short* __restrict__ Blo,
                                           int row, int nbase, int lmod, int grp, int tid,
                                           unsigned short* ldsB, v8f acc[8]) {
    const unsigned short* ah = Ahi + (size_t)row * DIM;
    const unsigned short* al = Alo + (size_t)row * DIM;

    stage_B(Bhi, Blo, ldsB, nbase, 0, tid);           // prime stage 0

#pragma unroll 1
    for (int s = 0; s < KSTEPS; ++s) {
        int kb = s * 32;
        int cur = s & 1;
        if (s + 1 < KSTEPS) {
            stage_B(Bhi, Blo, ldsB + (1 - cur) * BSTAGE, nbase, kb + 32, tid);
            wait_async_le<8>();                       // our previous-stage 8 loads done
        } else {
            wait_async_le<0>();
        }
        __syncthreads();                              // stage `cur` visible to all waves

        if (kb + 64 < DIM) __builtin_prefetch((const void*)(ah + kb + 64), 0, 0);
        // A fragment (lane holds row=lane%16; K=16*grp+0..15 contiguous)
        v16us a_hi = ld_frag16(ah + kb + 16 * grp, ah + kb + 16 * grp + 8);
        v16us a_lo = ld_frag16(al + kb + 16 * grp, al + kb + 16 * grp + 8);

        const unsigned short* lb = ldsB + cur * BSTAGE;
#pragma unroll
        for (int t = 0; t < 8; ++t) {
            const unsigned short* bh = lb + (t * 16 + lmod) * BROW + 16 * grp;
            const unsigned short* bl = bh + BPLANE;
            v16us b_hi = ld_frag16(bh, bh + 8);
            v16us b_lo = ld_frag16(bl, bl + 8);
            // bf16x3 fp32 emulation: hi*hi + lo*hi + hi*lo, all f32 accumulate
            acc[t] = wmma_bf16(a_hi, b_hi, acc[t]);
            acc[t] = wmma_bf16(a_lo, b_hi, acc[t]);
            acc[t] = wmma_bf16(a_hi, b_lo, acc[t]);
        }
        __syncthreads();                              // all reads done before buffer reuse
    }
}

// ---------- QKV GEMM + bias + scale, scatter into attention-friendly planes ----------
__global__ __launch_bounds__(128) void qkv_kernel(
    const unsigned short* __restrict__ xhi, const unsigned short* __restrict__ xlo,
    const unsigned short* __restrict__ whi, const unsigned short* __restrict__ wlo,
    const float* __restrict__ q_bias, const float* __restrict__ v_bias,
    unsigned short* __restrict__ qh, unsigned short* __restrict__ ql,
    unsigned short* __restrict__ kh, unsigned short* __restrict__ kl,
    unsigned short* __restrict__ vh, unsigned short* __restrict__ vl) {
    __shared__ unsigned short ldsB[2 * BSTAGE];

    int mblk = blockIdx.x % MB, ns = blockIdx.x / MB;  // MB*18 blocks
    int tid = threadIdx.x, wv = tid >> 5;
    int lane = tid & 31, lmod = lane & 15, grp = (lane >> 4) & 1;
    int mi = mblk * 4 + wv;
    int row = mi * 16 + lmod;

    v8f acc[8];
    const v8f vzero = {0.f, 0.f, 0.f, 0.f, 0.f, 0.f, 0.f, 0.f};
#pragma unroll
    for (int t = 0; t < 8; ++t) acc[t] = vzero;

    gemm_block(xhi, xlo, whi, wlo, row, ns * 128, lmod, grp, tid, ldsB, acc);

    int bArr[8], nnArr[8];
#pragma unroll
    for (int r = 0; r < 8; ++r) {
        int mg = mi * 16 + r + 8 * grp;                // C/D layout: M = r + 8*(lane/16)
        int bb = mg / NTOK;
        bArr[r] = bb; nnArr[r] = mg - bb * NTOK;
    }
#pragma unroll
    for (int t = 0; t < 8; ++t) {
        int n0 = ns * 128 + t * 16;
        int which = n0 / DIM;
        int rem = n0 - which * DIM;
        int h = rem >> 6;
        int d = (rem & 63) + lmod;                     // N = lane%16
        float bias = 0.f, scale = 1.f;
        if (which == 0) { bias = q_bias[h * HD + d]; scale = 0.125f; }  // 1/sqrt(64)
        else if (which == 2) { bias = v_bias[h * HD + d]; }
#pragma unroll
        for (int r = 0; r < 8; ++r) {
            float val = (acc[t][r] + bias) * scale;
            unsigned short hi_, lo_;
            splitf(val, hi_, lo_);
            size_t bh = (size_t)(bArr[r] * NH + h);
            if (which == 2) {                          // v stored transposed [d, j] for B-fragments of P@V
                size_t idx = (bh * HD + d) * VPAD + nnArr[r];
                vh[idx] = hi_; vl[idx] = lo_;
            } else {
                size_t idx = (bh * NPAD + nnArr[r]) * HD + d;
                if (which == 0) { qh[idx] = hi_; ql[idx] = lo_; }
                else            { kh[idx] = hi_; kl[idx] = lo_; }
            }
        }
    }
}

// ---------- fused attention: scores (bf16x3) + bias + softmax + P@V, one wave per 16-row q tile ----------
__global__ __launch_bounds__(128) void attn_kernel(
    const unsigned short* __restrict__ qh, const unsigned short* __restrict__ ql,
    const unsigned short* __restrict__ kh, const unsigned short* __restrict__ kl,
    const unsigned short* __restrict__ vh, const unsigned short* __restrict__ vl,
    const float* __restrict__ biasws,
    unsigned short* __restrict__ ohi, unsigned short* __restrict__ olo) {
    __shared__ unsigned short pbuf[4][16 * VPAD];      // per-wave P staging (C-layout -> A-layout transpose)

    int wv = threadIdx.x >> 5;
    int lane = threadIdx.x & 31, lmod = lane & 15, grp = (lane >> 4) & 1;
    int tile = blockIdx.x * 4 + wv;                    // B*H*13 = 9984 tiles
    int bh = tile / 13, rt = tile % 13;
    int b = bh / NH, h = bh % NH;

    size_t qkbase = (size_t)bh * NPAD * HD;
    int qrow = rt * 16 + lmod;

    const unsigned short* qph = qh + qkbase + (size_t)qrow * HD;
    const unsigned short* qpl = ql + qkbase + (size_t)qrow * HD;
    v16us aqh[2], aql[2];
#pragma unroll
    for (int s = 0; s < 2; ++s) {
        aqh[s] = ld_frag16(qph + s * 32 + 16 * grp, qph + s * 32 + 16 * grp + 8);
        aql[s] = ld_frag16(qpl + s * 32 + 16 * grp, qpl + s * 32 + 16 * grp + 8);
    }

    // ---- scores: 13 j-tiles of 16, K = 64 (2 slabs), bf16x3 ----
    v8f acc[13];
    const v8f vzero = {0.f, 0.f, 0.f, 0.f, 0.f, 0.f, 0.f, 0.f};
#pragma unroll
    for (int jt = 0; jt < 13; ++jt) acc[jt] = vzero;
#pragma unroll 1
    for (int jt = 0; jt < 13; ++jt) {
        int kj = jt * 16 + lmod;                       // B-fragment col = k-token j
        const unsigned short* kph = kh + qkbase + (size_t)kj * HD;
        const unsigned short* kpl = kl + qkbase + (size_t)kj * HD;
#pragma unroll
        for (int s = 0; s < 2; ++s) {
            const unsigned short* p0 = kph + s * 32 + 16 * grp;
            const unsigned short* p1 = kpl + s * 32 + 16 * grp;
            v16us b_hi = ld_frag16(p0, p0 + 8);
            v16us b_lo = ld_frag16(p1, p1 + 8);
            acc[jt] = wmma_bf16(aqh[s], b_hi, acc[jt]);
            acc[jt] = wmma_bf16(aql[s], b_hi, acc[jt]);
            acc[jt] = wmma_bf16(aqh[s], b_lo, acc[jt]);
        }
    }

    // ---- add relative-position bias, mask j >= 197 ----
    const float* bp = biasws + (size_t)h * NPAD * NPAD;
#pragma unroll
    for (int jt = 0; jt < 13; ++jt) {
        int j = jt * 16 + lmod;
        bool oob = (j >= NTOK);
#pragma unroll
        for (int r = 0; r < 8; ++r) {
            int i = rt * 16 + r + 8 * grp;
            float v = acc[jt][r] + bp[(size_t)i * NPAD + j];
            acc[jt][r] = oob ? -3.0e38f : v;
        }
    }

    // ---- softmax per row: reduce across 13 tiles + 16 lanes of the half-wave ----
    float rinv[8];
#pragma unroll
    for (int r = 0; r < 8; ++r) {
        float m = -3.0e38f;
#pragma unroll
        for (int jt = 0; jt < 13; ++jt) m = fmaxf(m, acc[jt][r]);
        m = fmaxf(m, __shfl_xor(m, 1));
        m = fmaxf(m, __shfl_xor(m, 2));
        m = fmaxf(m, __shfl_xor(m, 4));
        m = fmaxf(m, __shfl_xor(m, 8));
        float s = 0.f;
#pragma unroll
        for (int jt = 0; jt < 13; ++jt) {
            float e = __expf(acc[jt][r] - m);
            acc[jt][r] = e;
            s += e;
        }
        s += __shfl_xor(s, 1);
        s += __shfl_xor(s, 2);
        s += __shfl_xor(s, 4);
        s += __shfl_xor(s, 8);
        rinv[r] = 1.0f / s;
    }

    // ---- stage normalized P (bf16) to LDS: transpose C-layout -> A-layout ----
    unsigned short* pl = &pbuf[wv][0];
#pragma unroll
    for (int jt = 0; jt < 13; ++jt) {
        int j = jt * 16 + lmod;
#pragma unroll
        for (int r = 0; r < 8; ++r)
            pl[(r + 8 * grp) * VPAD + j] = bf16_rne(acc[jt][r] * rinv[r]);
    }
    {   // zero the K pad columns [208, 224)
        v4u z = {0u, 0u, 0u, 0u};
        *(v4u*)&pl[lmod * VPAD + 208 + grp * 8] = z;
    }
    __syncthreads();

    // ---- P @ V : K = 224 (7 slabs of 32), N = 64 (4 d-tiles), 2-term bf16 ----
    v8f oacc[4];
#pragma unroll
    for (int dt = 0; dt < 4; ++dt) oacc[dt] = vzero;
    size_t vbase = (size_t)bh * HD;
#pragma unroll 1
    for (int ks = 0; ks < 7; ++ks) {
        const unsigned short* pr = pl + lmod * VPAD + ks * 32;      // A-fragment of P from LDS
        v16us pa = ld_frag16(pr + 16 * grp, pr + 16 * grp + 8);
#pragma unroll
        for (int dt = 0; dt < 4; ++dt) {
            int d = dt * 16 + lmod;
            const unsigned short* vph = vh + (vbase + d) * VPAD + ks * 32 + 16 * grp;
            const unsigned short* vpl = vl + (vbase + d) * VPAD + ks * 32 + 16 * grp;
            v16us b_hi = ld_frag16(vph, vph + 8);
            v16us b_lo = ld_frag16(vpl, vpl + 8);
            oacc[dt] = wmma_bf16(pa, b_hi, oacc[dt]);
            oacc[dt] = wmma_bf16(pa, b_lo, oacc[dt]);
        }
    }

    // ---- store attention output as hi/lo planes for the projection GEMM ----
#pragma unroll
    for (int dt = 0; dt < 4; ++dt) {
        int d = dt * 16 + lmod;
#pragma unroll
        for (int r = 0; r < 8; ++r) {
            int i = rt * 16 + r + 8 * grp;
            if (i < NTOK) {
                size_t idx = ((size_t)(b * NTOK + i)) * DIM + h * HD + d;
                unsigned short hi_, lo_;
                splitf(oacc[dt][r], hi_, lo_);
                ohi[idx] = hi_; olo[idx] = lo_;
            }
        }
    }
}

// ---------- projection GEMM: out = o @ w_proj^T + b_proj ----------
__global__ __launch_bounds__(128) void proj_kernel(
    const unsigned short* __restrict__ ohi, const unsigned short* __restrict__ olo,
    const unsigned short* __restrict__ wph, const unsigned short* __restrict__ wpl,
    const float* __restrict__ b_proj, float* __restrict__ out) {
    __shared__ unsigned short ldsB[2 * BSTAGE];

    int mblk = blockIdx.x % MB, ns = blockIdx.x / MB;  // MB*6 blocks
    int tid = threadIdx.x, wv = tid >> 5;
    int lane = tid & 31, lmod = lane & 15, grp = (lane >> 4) & 1;
    int mi = mblk * 4 + wv;
    int row = mi * 16 + lmod;

    v8f acc[8];
    const v8f vzero = {0.f, 0.f, 0.f, 0.f, 0.f, 0.f, 0.f, 0.f};
#pragma unroll
    for (int t = 0; t < 8; ++t) acc[t] = vzero;

    gemm_block(ohi, olo, wph, wpl, row, ns * 128, lmod, grp, tid, ldsB, acc);

#pragma unroll
    for (int t = 0; t < 8; ++t) {
        int n = ns * 128 + t * 16 + lmod;
        float bb = b_proj[n];
#pragma unroll
        for (int r = 0; r < 8; ++r) {
            int mg = mi * 16 + r + 8 * grp;
            out[(size_t)mg * DIM + n] = acc[t][r] + bb;
        }
    }
}

extern "C" void kernel_launch(void* const* d_in, const int* in_sizes, int n_in,
                              void* d_out, int out_size, void* d_ws, size_t ws_size,
                              hipStream_t stream) {
    (void)in_sizes; (void)n_in; (void)out_size; (void)ws_size;
    const float* x         = (const float*)d_in[0];
    const float* w_qkv     = (const float*)d_in[1];
    const float* q_bias    = (const float*)d_in[2];
    const float* v_bias    = (const float*)d_in[3];
    const float* rel_table = (const float*)d_in[4];
    const float* w_proj    = (const float*)d_in[5];
    const float* b_proj    = (const float*)d_in[6];
    const int*   rel_index = (const int*)d_in[7];
    float* out = (float*)d_out;

    char* p = (char*)d_ws;
    auto carve = [&](size_t bytes) -> char* {
        char* r = p;
        p += (bytes + 255) & ~(size_t)255;
        return r;
    };
    unsigned short* xhi = (unsigned short*)carve((size_t)MROWS * DIM * 2);
    unsigned short* xlo = (unsigned short*)carve((size_t)MROWS * DIM * 2);
    unsigned short* wqh = (unsigned short*)carve((size_t)3 * DIM * DIM * 2);
    unsigned short* wql = (unsigned short*)carve((size_t)3 * DIM * DIM * 2);
    unsigned short* wph = (unsigned short*)carve((size_t)DIM * DIM * 2);
    unsigned short* wpl = (unsigned short*)carve((size_t)DIM * DIM * 2);
    unsigned short* qhp = (unsigned short*)carve((size_t)BH * NPAD * HD * 2);
    unsigned short* qlp = (unsigned short*)carve((size_t)BH * NPAD * HD * 2);
    unsigned short* khp = (unsigned short*)carve((size_t)BH * NPAD * HD * 2);
    unsigned short* klp = (unsigned short*)carve((size_t)BH * NPAD * HD * 2);
    unsigned short* vhp = (unsigned short*)carve((size_t)BH * HD * VPAD * 2);
    unsigned short* vlp = (unsigned short*)carve((size_t)BH * HD * VPAD * 2);
    unsigned short* ohp = (unsigned short*)carve((size_t)MROWS * DIM * 2);
    unsigned short* olp = (unsigned short*)carve((size_t)MROWS * DIM * 2);
    float* biasws = (float*)carve((size_t)NH * NPAD * NPAD * 4);

    // 1) split fp32 operands into bf16 hi/lo planes (GEMMs then do zero conversion work)
    split_kernel<<<(MROWS * DIM + 255) / 256, 256, 0, stream>>>(x, xhi, xlo, MROWS * DIM);
    split_kernel<<<(3 * DIM * DIM + 255) / 256, 256, 0, stream>>>(w_qkv, wqh, wql, 3 * DIM * DIM);
    split_kernel<<<(DIM * DIM + 255) / 256, 256, 0, stream>>>(w_proj, wph, wpl, DIM * DIM);
    // 2) relative position bias gather -> dense [H][208][208]
    biasfill_kernel<<<(NTOK * NTOK + 255) / 256, 256, 0, stream>>>(rel_table, rel_index, biasws);
    // 3) QKV GEMM (bf16x3, async-LDS staged B strip) with fused bias/scale
    qkv_kernel<<<MB * 18, 128, 0, stream>>>(xhi, xlo, wqh, wql, q_bias, v_bias,
                                            qhp, qlp, khp, klp, vhp, vlp);
    // 4) fused attention (scores + bias + softmax + P@V), 1 wave per 16-row tile
    attn_kernel<<<(BH * 13) / 4, 128, 0, stream>>>(qhp, qlp, khp, klp, vhp, vlp, biasws, ohp, olp);
    // 5) projection GEMM (bf16x3, async-LDS staged B strip) + bias -> fp32 output
    proj_kernel<<<MB * 6, 128, 0, stream>>>(ohp, olp, wph, wpl, b_proj, out);
}
